// RNNDecoder_35820027248848
// MI455X (gfx1250) — compile-verified, matrix operand order checked by reading, not run on hardware
//
#include <hip/hip_runtime.h>
#include <hip/hip_bf16.h>
#include <math.h>

// Problem constants (from reference)
#define V_SZ  32000
#define E_SZ  512
#define H_SZ  512
#define L_SZ  2
#define B_SZ  64
#define T_SZ  65
#define TM1   64          // T-1 decode steps
#define ROWS  (B_SZ*TM1)  // 4096 GEMM rows for gi0_all / ys / logits
#define G3H   (3*H_SZ)    // 1536

typedef __bf16 bf16;
typedef __attribute__((ext_vector_type(8)))  __bf16 v8bf;
typedef __attribute__((ext_vector_type(16))) __bf16 v16bf;
typedef __attribute__((ext_vector_type(8)))  float  v8f;
typedef int v4i_ __attribute__((vector_size(16)));   // matches async-LDS builtin proto

#define AS_GLOBAL __attribute__((address_space(1)))
#define AS_LDS    __attribute__((address_space(3)))

#define DEVI __device__ __forceinline__

// ---------------------------------------------------------------------------
// WMMA helpers (CDNA5 wave32, V_WMMA_F32_16X16X32_BF16)
// A-frag (16x32 MxK) and B-frag-from-transposed-storage (N x K row-major)
// use the identical addressing: lane = row%16, g = lane/16,
// elems 0..7 = [k + g*8 .. +7], elems 8..15 = [k + 16 + g*8 .. +7].
// ---------------------------------------------------------------------------
DEVI v16bf frag_g(const bf16* __restrict__ base, int ld, int row, int k, int lane) {
  int r = row + (lane & 15);
  int g = lane >> 4;
  const bf16* p = base + (size_t)r * ld + k + g * 8;
  v8bf lo = *(const v8bf*)(p);
  v8bf hi = *(const v8bf*)(p + 16);
  v16bf f;
#pragma unroll
  for (int i = 0; i < 8; ++i) { f[i] = lo[i]; f[8 + i] = hi[i]; }
  return f;
}

DEVI v8f wmma_bf16(v16bf a, v16bf b, v8f c) {
  return __builtin_amdgcn_wmma_f32_16x16x32_bf16(
      /*neg_a=*/false, a, /*neg_b=*/false, b,
      /*c_mod=*/(short)0, c, /*reuse_a=*/false, /*reuse_b=*/false);
}

// ---------------------------------------------------------------------------
// Generic bf16 GEMM:  C[MxN] (f32) = A[MxK] @ B[NxK]^T + bias[N]
// block = 256 threads = 8 waves (2 in M x 4 in N) -> 64x128 output per block.
// grid = (N/128, M/64). Direct global fragment loads (A is tiny / L2-hot).
// ---------------------------------------------------------------------------
__global__ __launch_bounds__(256)
void gemm_bf16(const bf16* __restrict__ A, const bf16* __restrict__ Bm,
               const float* __restrict__ bias, float* __restrict__ C,
               int M, int N, int K) {
  const int lane = threadIdx.x & 31;
  const int wave = threadIdx.x >> 5;
  const int mbase = blockIdx.y * 64 + (wave & 1) * 32;
  const int nbase = blockIdx.x * 128 + (wave >> 1) * 32;

  v8f acc00 = {}, acc01 = {}, acc10 = {}, acc11 = {};
  for (int k = 0; k < K; k += 32) {
    v16bf a0 = frag_g(A,  K, mbase +  0, k, lane);
    v16bf a1 = frag_g(A,  K, mbase + 16, k, lane);
    v16bf b0 = frag_g(Bm, K, nbase +  0, k, lane);
    v16bf b1 = frag_g(Bm, K, nbase + 16, k, lane);
    acc00 = wmma_bf16(a0, b0, acc00);
    acc01 = wmma_bf16(a0, b1, acc01);
    acc10 = wmma_bf16(a1, b0, acc10);
    acc11 = wmma_bf16(a1, b1, acc11);
  }

  const int g  = lane >> 4;
  const int n0 = lane & 15;
#pragma unroll
  for (int j = 0; j < 2; ++j) {
    int col = nbase + j * 16 + n0;
    float bv = bias ? bias[col] : 0.0f;
    v8f aj0 = (j == 0) ? acc00 : acc01;
    v8f aj1 = (j == 0) ? acc10 : acc11;
#pragma unroll
    for (int v = 0; v < 8; ++v) {
      int r0 = mbase + g * 8 + v;
      int r1 = mbase + 16 + g * 8 + v;
      C[(size_t)r0 * N + col] = aj0[v] + bv;
      C[(size_t)r1 * N + col] = aj1[v] + bv;
    }
  }
}

// ---------------------------------------------------------------------------
// FC GEMM: logits[4096 x 32000] = ys[4096x512] @ W_fc[32000x512]^T + b_fc.
// A-panel (64x512 bf16 = 64KB) staged in LDS via GLOBAL_LOAD_ASYNC_TO_LDS_B128
// (ASYNCcnt-tracked, no VGPR round-trip) when the toolchain exposes it;
// W_fc streams through the 192MB L2.
// 8 waves arranged 2(M) x 4(N); each wave owns a 32x64 register tile
// (2x4 WMMA tiles -> 8 WMMAs per K=32 slice from 6 fragment loads).
// Block covers 64 x 256 of output; grid = (32000/256, 4096/64).
// ---------------------------------------------------------------------------
__global__ __launch_bounds__(256)
void gemm_fc(const bf16* __restrict__ A, const bf16* __restrict__ Bm,
             const float* __restrict__ bias, float* __restrict__ C) {
  __shared__ bf16 As[64 * 512];  // 64 KB of the 320 KB WGP LDS

  const int mblk = blockIdx.y * 64;
  const int nblk = blockIdx.x * 256;

  // ---- stage the A panel into LDS -------------------------------------
#if __has_builtin(__builtin_amdgcn_global_load_async_to_lds_b128) && \
    __has_builtin(__builtin_amdgcn_s_wait_asynccnt)
  {
    const bf16* gbase = A + (size_t)mblk * 512;
    for (int i = threadIdx.x; i < 4096; i += 256)   // 4096 x 16B chunks
      __builtin_amdgcn_global_load_async_to_lds_b128(
          (AS_GLOBAL v4i_*)(gbase + (size_t)i * 8),
          (AS_LDS v4i_*)(As + i * 8),
          0, 0);
    __builtin_amdgcn_s_wait_asynccnt(0);
  }
#else
  {
    const uint4* gsrc = (const uint4*)(A + (size_t)mblk * 512);
    uint4* ldst = (uint4*)As;
    for (int i = threadIdx.x; i < 4096; i += 256) ldst[i] = gsrc[i];
  }
#endif
  __syncthreads();

  const int lane = threadIdx.x & 31;
  const int wave = threadIdx.x >> 5;
  const int mo = (wave & 1) * 32;            // M offset within panel
  const int nb = nblk + (wave >> 1) * 64;    // N base for this wave
  const int g  = lane >> 4;
  const int r0 = mo + (lane & 15);

  v8f acc[2][4];
#pragma unroll
  for (int i = 0; i < 2; ++i)
#pragma unroll
    for (int j = 0; j < 4; ++j) acc[i][j] = (v8f){};

  for (int k = 0; k < 512; k += 32) {
    // A fragments from LDS
    v16bf a[2];
#pragma unroll
    for (int i = 0; i < 2; ++i) {
      const bf16* p = &As[(r0 + i * 16) * 512 + k + g * 8];
      v8bf lo = *(const v8bf*)(p);
      v8bf hi = *(const v8bf*)(p + 16);
#pragma unroll
      for (int e = 0; e < 8; ++e) { a[i][e] = lo[e]; a[i][8 + e] = hi[e]; }
    }
    // prefetch next K-slice of B while we compute
    if (k + 32 < 512)
      __builtin_prefetch(Bm + (size_t)(nb + (lane & 15)) * 512 + k + 32, 0, 1);
    // B fragments from global (L2-resident W_fc)
    v16bf b[4];
#pragma unroll
    for (int j = 0; j < 4; ++j) b[j] = frag_g(Bm, 512, nb + j * 16, k, lane);
#pragma unroll
    for (int i = 0; i < 2; ++i)
#pragma unroll
      for (int j = 0; j < 4; ++j)
        acc[i][j] = wmma_bf16(a[i], b[j], acc[i][j]);
  }

  const int n0 = lane & 15;
#pragma unroll
  for (int j = 0; j < 4; ++j) {
    int col = nb + j * 16 + n0;
    float bv = bias[col];
#pragma unroll
    for (int i = 0; i < 2; ++i) {
#pragma unroll
      for (int v = 0; v < 8; ++v) {
        int r = mblk + mo + i * 16 + g * 8 + v;
        C[(size_t)r * V_SZ + col] = acc[i][j][v] + bv;
      }
    }
  }
}

// ---------------------------------------------------------------------------
// Small elementwise / setup kernels
// ---------------------------------------------------------------------------
__global__ void k_f32_to_bf16(const float* __restrict__ in, bf16* __restrict__ out, int n) {
  for (int i = blockIdx.x * blockDim.x + threadIdx.x; i < n; i += gridDim.x * blockDim.x)
    out[i] = (bf16)in[i];
}

// Build teacher-forced input rows (t-major): Xall[t*B+b] = emb[tok(b,t)], bf16.
__global__ void k_embed_gather(const int* __restrict__ targets,
                               const float* __restrict__ emb,
                               bf16* __restrict__ Xall) {
  int row = blockIdx.x;            // 0..4095, row = t*B + b
  int t = row >> 6, b = row & 63;
  int tok = (t == 0) ? 1 : targets[b * T_SZ + (t - 1)];
  const float* e = emb + (size_t)tok * E_SZ;
  bf16* o = Xall + (size_t)row * E_SZ;
  for (int j = threadIdx.x; j < E_SZ; j += blockDim.x) o[j] = (bf16)e[j];
}

// h0 = tanh(context @ W_init^T + b_init); layer 0 -> hA, layer 1 -> hB.
__global__ void k_init_hidden(const float* __restrict__ ctx,
                              const float* __restrict__ Wini,
                              const float* __restrict__ bini,
                              float* __restrict__ hA, float* __restrict__ hB,
                              bf16* __restrict__ hAb, bf16* __restrict__ hBb) {
  int r = blockIdx.x;              // 0..127 = l*64 + b
  int l = r >> 6, b = r & 63;
  const float* x = ctx + (size_t)r * H_SZ;
  for (int j = threadIdx.x; j < H_SZ; j += blockDim.x) {
    const float* w = Wini + (size_t)j * H_SZ;
    float s = 0.0f;
    for (int k = 0; k < H_SZ; ++k) s = fmaf(x[k], w[k], s);
    float v = tanhf(s + bini[j]);
    if (l == 0) { hA[b * H_SZ + j] = v; hAb[b * H_SZ + j] = (bf16)v; }
    else        { hB[b * H_SZ + j] = v; hBb[b * H_SZ + j] = (bf16)v; }
  }
}

DEVI float sigmoidf_(float x) { return 1.0f / (1.0f + expf(-x)); }

// GRU gate combine (PyTorch order r,z,n). gi/gh rows are 3H wide, bias already
// folded in by the GEMMs. Updates h in place and its bf16 mirror.
__global__ void k_gates(const float* __restrict__ gi, const float* __restrict__ gh,
                        float* __restrict__ h, bf16* __restrict__ hb,
                        bf16* __restrict__ ysb, int t) {
  int idx = blockIdx.x * blockDim.x + threadIdx.x;  // 0..B*H-1
  int b = idx >> 9, j = idx & 511;
  const float* gib = gi + (size_t)b * G3H;
  const float* ghb = gh + (size_t)b * G3H;
  float r = sigmoidf_(gib[j] + ghb[j]);
  float z = sigmoidf_(gib[H_SZ + j] + ghb[H_SZ + j]);
  float n = tanhf(gib[2 * H_SZ + j] + r * ghb[2 * H_SZ + j]);
  float hv = h[idx];
  float o = (1.0f - z) * n + z * hv;
  h[idx] = o;
  hb[idx] = (bf16)o;
  if (ysb) ysb[((size_t)b * TM1 + t) * H_SZ + j] = (bf16)o;  // b-major for FC
}

// In-place log-softmax over V per row (4096 rows); row (128KB) is L2-hot
// between passes.
__global__ __launch_bounds__(256)
void k_log_softmax(float* __restrict__ out) {
  __shared__ float red[256];
  float* p = out + (size_t)blockIdx.x * V_SZ;
  int tid = threadIdx.x;

  float m = -INFINITY;
  for (int i = tid; i < V_SZ; i += 256) m = fmaxf(m, p[i]);
  red[tid] = m; __syncthreads();
  for (int s = 128; s > 0; s >>= 1) {
    if (tid < s) red[tid] = fmaxf(red[tid], red[tid + s]);
    __syncthreads();
  }
  float rowmax = red[0]; __syncthreads();

  float sum = 0.0f;
  for (int i = tid; i < V_SZ; i += 256) sum += expf(p[i] - rowmax);
  red[tid] = sum; __syncthreads();
  for (int s = 128; s > 0; s >>= 1) {
    if (tid < s) red[tid] += red[tid + s];
    __syncthreads();
  }
  float lse = rowmax + logf(red[0]);

  for (int i = tid; i < V_SZ; i += 256) p[i] -= lse;
}

// ---------------------------------------------------------------------------
// Host orchestration
// ---------------------------------------------------------------------------
extern "C" void kernel_launch(void* const* d_in, const int* in_sizes, int n_in,
                              void* d_out, int out_size, void* d_ws, size_t ws_size,
                              hipStream_t stream) {
  const float* ctx     = (const float*)d_in[0];
  const int*   targets = (const int*)  d_in[1];
  const float* emb     = (const float*)d_in[2];
  const float* W_init  = (const float*)d_in[3];
  const float* b_init  = (const float*)d_in[4];
  const float* Wi0     = (const float*)d_in[5];
  const float* Wh0     = (const float*)d_in[6];
  const float* bi0     = (const float*)d_in[7];
  const float* bh0     = (const float*)d_in[8];
  const float* Wi1     = (const float*)d_in[9];
  const float* Wh1     = (const float*)d_in[10];
  const float* bi1     = (const float*)d_in[11];
  const float* bh1     = (const float*)d_in[12];
  const float* W_fc    = (const float*)d_in[13];
  const float* b_fc    = (const float*)d_in[14];
  float* out = (float*)d_out;

  // workspace carve (256B aligned); ~74 MB total
  char* base = (char*)d_ws;
  size_t off = 0;
  auto carve = [&](size_t bytes) -> char* {
    char* p = base + off;
    off = (off + bytes + 255) & ~(size_t)255;
    return p;
  };
  bf16*  Wi0b   = (bf16*)carve((size_t)G3H * E_SZ * 2);
  bf16*  Wh0b   = (bf16*)carve((size_t)G3H * H_SZ * 2);
  bf16*  Wi1b   = (bf16*)carve((size_t)G3H * H_SZ * 2);
  bf16*  Wh1b   = (bf16*)carve((size_t)G3H * H_SZ * 2);
  bf16*  Wfcb   = (bf16*)carve((size_t)V_SZ * H_SZ * 2);
  bf16*  Xallb  = (bf16*)carve((size_t)ROWS * E_SZ * 2);
  bf16*  ysb    = (bf16*)carve((size_t)ROWS * H_SZ * 2);
  float* gi0all = (float*)carve((size_t)ROWS * G3H * 4);
  float* hA     = (float*)carve((size_t)B_SZ * H_SZ * 4);
  float* hB     = (float*)carve((size_t)B_SZ * H_SZ * 4);
  bf16*  hAb    = (bf16*)carve((size_t)B_SZ * H_SZ * 2);
  bf16*  hBb    = (bf16*)carve((size_t)B_SZ * H_SZ * 2);
  float* gh0    = (float*)carve((size_t)B_SZ * G3H * 4);
  float* gi1    = (float*)carve((size_t)B_SZ * G3H * 4);
  float* gh1    = (float*)carve((size_t)B_SZ * G3H * 4);
  (void)ws_size; (void)n_in; (void)in_sizes; (void)out_size;

  // 1) weight conversions to bf16
  k_f32_to_bf16<<<512, 256, 0, stream>>>(Wi0,  Wi0b, G3H * E_SZ);
  k_f32_to_bf16<<<512, 256, 0, stream>>>(Wh0,  Wh0b, G3H * H_SZ);
  k_f32_to_bf16<<<512, 256, 0, stream>>>(Wi1,  Wi1b, G3H * H_SZ);
  k_f32_to_bf16<<<512, 256, 0, stream>>>(Wh1,  Wh1b, G3H * H_SZ);
  k_f32_to_bf16<<<2048, 256, 0, stream>>>(W_fc, Wfcb, V_SZ * H_SZ);

  // 2) teacher-forced embedding gather (t-major rows) + initial hidden
  k_embed_gather<<<ROWS, 256, 0, stream>>>(targets, emb, Xallb);
  k_init_hidden<<<L_SZ * B_SZ, 256, 0, stream>>>(ctx, W_init, b_init, hA, hB, hAb, hBb);

  // 3) hoisted input-gate GEMM for layer 0 over all timesteps:
  //    gi0all[4096 x 1536] = Xall @ Wi0^T + bi0
  gemm_bf16<<<dim3(G3H / 128, ROWS / 64), 256, 0, stream>>>(
      Xallb, Wi0b, bi0, gi0all, ROWS, G3H, E_SZ);

  // 4) sequential recurrence, 64 steps; per-step launches enforce ordering
  for (int t = 0; t < TM1; ++t) {
    gemm_bf16<<<dim3(G3H / 128, 1), 256, 0, stream>>>(
        hAb, Wh0b, bh0, gh0, B_SZ, G3H, H_SZ);
    k_gates<<<(B_SZ * H_SZ) / 256, 256, 0, stream>>>(
        gi0all + (size_t)t * B_SZ * G3H, gh0, hA, hAb, (bf16*)nullptr, t);
    gemm_bf16<<<dim3(G3H / 128, 1), 256, 0, stream>>>(
        hAb, Wi1b, bi1, gi1, B_SZ, G3H, H_SZ);
    gemm_bf16<<<dim3(G3H / 128, 1), 256, 0, stream>>>(
        hBb, Wh1b, bh1, gh1, B_SZ, G3H, H_SZ);
    k_gates<<<(B_SZ * H_SZ) / 256, 256, 0, stream>>>(
        gi1, gh1, hB, hBb, ysb, t);
  }

  // 5) FC projection (dominant GEMM, 32x64 per-wave tile) + log-softmax
  gemm_fc<<<dim3(V_SZ / 256, ROWS / 64), 256, 0, stream>>>(ysb, Wfcb, b_fc, out);
  k_log_softmax<<<ROWS, 256, 0, stream>>>(out);
}
